// MSAAttenBlock_70660801953911
// MI455X (gfx1250) — matrix operand order repaired and checked by reading, not compiled, not run on hardware
//
#include <hip/hip_runtime.h>

#define DEVI __device__ __forceinline__

typedef __attribute__((ext_vector_type(16))) _Float16 v16h;
typedef __attribute__((ext_vector_type(8)))  float    v8f;

constexpr int S = 128, R = 256, CM = 256, H = 8, CA = 32, CZ = 128;
constexpr int HC = H * CA;          // 256
constexpr int CM4 = 4 * CM;         // 1024
constexpr int SR = S * R;           // 32768

// ---------------------------------------------------------------------------
// WMMA helpers (CDNA5: D = A(16x32 f16) x B(32x16 f16) + C(16x16 f32))
// ---------------------------------------------------------------------------
DEVI v8f wmma16(v16h a, v16h b, v8f c) {
  return __builtin_amdgcn_wmma_f32_16x16x32_f16(
      /*neg_a=*/false, a, /*neg_b=*/false, b,
      /*c_mod=*/(short)0, c, /*reuse_a=*/false, /*reuse_b=*/false);
}

union AF { v16h v; uint4 u[2]; };

// A operand, row-major [16][>=32] source: lane m (0-15) holds row m,
// halfs 0..7 = K kb..kb+7, halfs 8..15 = K kb+16..kb+23 (kb = 0 / 8).
DEVI v16h load_a_frag(const _Float16* base, int stride) {
  const int l = threadIdx.x & 31;
  const _Float16* p = base + (l & 15) * stride + ((l < 16) ? 0 : 8);
  AF f;
  f.u[0] = *(const uint4*)p;
  f.u[1] = *(const uint4*)(p + 16);
  return f.v;
}

// B operand from B^T layout [n][k]: lane n (0-15) holds column n,
// halfs 0..7 = K kb..kb+7, halfs 8..15 = K kb+16..kb+23 (kb = 0 / 8).
DEVI v16h load_b_frag(const _Float16* baseT, int stride) {
  const int l = threadIdx.x & 31;
  const _Float16* p = baseT + (l & 15) * stride + ((l < 16) ? 0 : 8);
  AF f;
  f.u[0] = *(const uint4*)p;
  f.u[1] = *(const uint4*)(p + 16);
  return f.v;
}

// CDNA5 async global->LDS DMA: copy 64B per lane (4 x b128), ASYNCcnt-tracked.
// Same INST_OFFSET advances both LDS destination and global source.
DEVI void async_cp64(unsigned lds_off, const _Float16* gsrc) {
  asm volatile(
      "global_load_async_to_lds_b128 %0, %1, off\n\t"
      "global_load_async_to_lds_b128 %0, %1, off offset:16\n\t"
      "global_load_async_to_lds_b128 %0, %1, off offset:32\n\t"
      "global_load_async_to_lds_b128 %0, %1, off offset:48"
      :: "v"(lds_off), "v"(gsrc) : "memory");
}
DEVI void wait_async0() { asm volatile("s_wait_asynccnt 0x0" ::: "memory"); }
DEVI unsigned lds_off32(const void* p) { return (unsigned)(uintptr_t)p; }

// ---------------------------------------------------------------------------
// Tiled WMMA GEMM: C[M,N] = act(alpha * A[M,K] @ Bt^T + bias)
// A is [M,K] f16 row-major; Bt is the WEIGHT PRE-TRANSPOSED to [N,K] f16,
// so both tiles stage as 128 rows x 64 contiguous halfs via async DMA
// (no VGPR staging, no LDS transpose, DMA overlaps WMMA of current buffer).
// ACT: 0 none, 1 sigmoid, 2 relu.  OUT_F16: write f16, else f32 (ACC: +=).
// ---------------------------------------------------------------------------
template <int ACT, bool OUT_F16, bool ACC, bool HAS_BIAS>
__global__ __launch_bounds__(256) void gemm_kernel(
    const _Float16* __restrict__ A, const _Float16* __restrict__ Bt,
    const float* __restrict__ bias, float* __restrict__ Cf,
    _Float16* __restrict__ Ch, int M, int N, int K, float alpha) {
  __shared__ _Float16 As[2][128][72];   // [buf][m][k], pad 8 halfs
  __shared__ _Float16 Bs[2][128][72];   // [buf][n][k]
  const int tid = threadIdx.x;
  const int wave = tid >> 5, wm = wave >> 1, wn = wave & 1;
  const int m0 = blockIdx.y * 128, n0 = blockIdx.x * 128;

  const int sr = tid >> 1, ssg = (tid & 1) << 5;   // stager: row, 32-half seg
  const _Float16* Ag = A  + (size_t)(m0 + sr) * K + ssg;
  const _Float16* Bg = Bt + (size_t)(n0 + sr) * K + ssg;

  auto issue = [&](int k0, int buf) {
    async_cp64(lds_off32(&As[buf][sr][ssg]), Ag + k0);
    async_cp64(lds_off32(&Bs[buf][sr][ssg]), Bg + k0);
  };

  v8f acc[2][4] = {};
  const int NIT = K >> 6;
  issue(0, 0);
  for (int it = 0; it < NIT; ++it) {
    const int cur = it & 1;
    wait_async0();        // my DMA into buf[cur] done
    __syncthreads();      // everyone's DMA done; buf[cur^1] free to overwrite
    if (it + 1 < NIT) issue((it + 1) << 6, cur ^ 1);
#pragma unroll
    for (int ks = 0; ks < 2; ++ks) {
      v16h af[2], bf[4];
#pragma unroll
      for (int mt = 0; mt < 2; ++mt)
        af[mt] = load_a_frag(&As[cur][wm * 32 + mt * 16][ks * 32], 72);
#pragma unroll
      for (int nt = 0; nt < 4; ++nt)
        bf[nt] = load_b_frag(&Bs[cur][wn * 64 + nt * 16][ks * 32], 72);
#pragma unroll
      for (int mt = 0; mt < 2; ++mt)
#pragma unroll
        for (int nt = 0; nt < 4; ++nt)
          acc[mt][nt] = wmma16(af[mt], bf[nt], acc[mt][nt]);
    }
  }

  // epilogue: C layout -> VGPR i holds (M = i + 8*(lane>=16), N = lane&15)
  const int l = tid & 31, lro = (l < 16) ? 0 : 8, lc = l & 15;
  float bv[4];
#pragma unroll
  for (int nt = 0; nt < 4; ++nt)
    bv[nt] = HAS_BIAS ? bias[n0 + wn * 64 + nt * 16 + lc] : 0.0f;
#pragma unroll
  for (int mt = 0; mt < 2; ++mt)
#pragma unroll
    for (int nt = 0; nt < 4; ++nt) {
      const int row0 = m0 + wm * 32 + mt * 16 + lro;
      const int col = n0 + wn * 64 + nt * 16 + lc;
      float res[8];
#pragma unroll
      for (int i = 0; i < 8; ++i) {
        float v = acc[mt][nt][i] * alpha + bv[nt];
        if (ACT == 1) v = 1.0f / (1.0f + __expf(-v));
        else if (ACT == 2) v = fmaxf(v, 0.0f);
        res[i] = v;
      }
      if (OUT_F16) {
        _Float16* p = Ch + (size_t)row0 * N + col;
#pragma unroll
        for (int i = 0; i < 8; ++i) p[(size_t)i * N] = (_Float16)res[i];
      } else if (ACC) {
        float* p = Cf + (size_t)row0 * N + col;
        float old[8];
#pragma unroll
        for (int i = 0; i < 8; ++i) old[i] = p[(size_t)i * N];  // batched loads
#pragma unroll
        for (int i = 0; i < 8; ++i) p[(size_t)i * N] = old[i] + res[i];
      } else {
        float* p = Cf + (size_t)row0 * N + col;
#pragma unroll
        for (int i = 0; i < 8; ++i) p[(size_t)i * N] = res[i];
      }
    }
}

// ---------------------------------------------------------------------------
// Fused attention: one block per (outer, head).
// Row (IS_COL=false): outer = s, attend over L = R keys, pair bias + mask.
// Col (IS_COL=true):  outer = r, attend over L = S keys, mask only.
// 8 waves; each wave processes 16 query rows per pass (row: 2 passes).
// ---------------------------------------------------------------------------
template <bool IS_COL>
__global__ __launch_bounds__(256) void atten_kernel(
    const _Float16* __restrict__ q, const _Float16* __restrict__ k,
    const _Float16* __restrict__ v, const _Float16* __restrict__ g,
    const float* __restrict__ pb, const float* __restrict__ mask,
    _Float16* __restrict__ o) {
  constexpr int L = IS_COL ? S : R;   // query/key length
  constexpr int NT = L / 16;          // logits n-tiles
  constexpr int KC = L / 32;          // PV k-chunks
  constexpr int PASSES = L / 128;     // 16 rows per wave per pass
  __shared__ _Float16 Ksh[L][40];           // K rows [key][c], pad 8
  __shared__ _Float16 Vt[32][L + 8];        // V transposed [c][key]
  __shared__ _Float16 Psh[8][16][L + 8];    // per-wave softmax rows
  const int tid = threadIdx.x, wave = tid >> 5;
  const int outer = blockIdx.x, h = blockIdx.y;
  size_t base; int rstride; const float* mrow; int mstride;
  if (IS_COL) {
    base = (size_t)outer * HC + h * CA;        rstride = R * HC;
    mrow = mask + outer;                       mstride = R;       // m_mask[key_s][r]
  } else {
    base = (size_t)outer * R * HC + h * CA;    rstride = HC;
    mrow = mask + (size_t)outer * R;           mstride = 1;       // m_mask[s][key_r]
  }
  const float* pbh = IS_COL ? nullptr : (pb + (size_t)h * R * R);

  for (int r = tid; r < L; r += 256)                 // stage K via async DMA
    async_cp64(lds_off32(&Ksh[r][0]), k + base + (size_t)r * rstride);
  for (int idx = tid; idx < L * 32; idx += 256) {    // stage V^T (transpose)
    const int r = idx >> 5, c = idx & 31;
    Vt[c][r] = v[base + (size_t)r * rstride + c];
  }
  wait_async0();
  __syncthreads();

  const int l = tid & 31, lro = (l < 16) ? 0 : 8, lc = l & 15;
  for (int pass = 0; pass < PASSES; ++pass) {
    const int qbase = (wave * PASSES + pass) * 16;
    // Q fragment straight from global, already in WMMA A layout
    AF qa;
    {
      const _Float16* qp = q + base + (size_t)(qbase + lc) * rstride + ((l < 16) ? 0 : 8);
      qa.u[0] = *(const uint4*)qp;
      qa.u[1] = *(const uint4*)(qp + 16);
    }
    // logits: Q[16x32] x K^T -> NT tiles of 16x16 (CA=32 = one WMMA k-step)
    v8f logit[NT];
#pragma unroll
    for (int t = 0; t < NT; ++t) {
      v8f z{};
      logit[t] = wmma16(qa.v, load_b_frag(&Ksh[t * 16][0], 40), z);
    }
    // + mask bias (+ pair bias for row attention)
#pragma unroll
    for (int t = 0; t < NT; ++t) {
      const int key = t * 16 + lc;
      const float mb = (mrow[key * mstride] - 1.0f) * 1e9f;
#pragma unroll
      for (int i = 0; i < 8; ++i) {
        float add = mb;
        if (!IS_COL) add += pbh[(size_t)(qbase + i + lro) * R + key];
        logit[t][i] += add;
      }
    }
    // softmax over keys: per-lane over NT regs, then 16-lane xor-shuffle
#pragma unroll
    for (int i = 0; i < 8; ++i) {
      float mx = -1e30f;
#pragma unroll
      for (int t = 0; t < NT; ++t) mx = fmaxf(mx, logit[t][i]);
#pragma unroll
      for (int d = 1; d < 16; d <<= 1) mx = fmaxf(mx, __shfl_xor(mx, d, 32));
      float sum = 0.0f;
#pragma unroll
      for (int t = 0; t < NT; ++t) {
        const float e = __expf(logit[t][i] - mx);
        logit[t][i] = e;
        sum += e;
      }
#pragma unroll
      for (int d = 1; d < 16; d <<= 1) sum += __shfl_xor(sum, d, 32);
      const float inv = 1.0f / sum;
#pragma unroll
      for (int t = 0; t < NT; ++t)
        Psh[wave][i + lro][t * 16 + lc] = (_Float16)(logit[t][i] * inv);
    }
    asm volatile("" ::: "memory");  // same-wave LDS ops complete in order
    // O[16x32] = P[16xL] x V[Lx32]
    v8f oacc[2] = {};
#pragma unroll
    for (int kc = 0; kc < KC; ++kc) {
      const v16h pa = load_a_frag(&Psh[wave][0][kc * 32], L + 8);
#pragma unroll
      for (int nt = 0; nt < 2; ++nt)
        oacc[nt] = wmma16(pa, load_b_frag(&Vt[nt * 16][kc * 32], L + 8), oacc[nt]);
    }
    // gated write: o = g * (P V)
#pragma unroll
    for (int nt = 0; nt < 2; ++nt)
#pragma unroll
      for (int i = 0; i < 8; ++i) {
        const size_t idx = base + (size_t)(qbase + i + lro) * rstride + nt * 16 + lc;
        o[idx] = (_Float16)((float)g[idx] * oacc[nt][i]);
      }
  }
}

// ---------------------------------------------------------------------------
// LayerNorm over CM=256, one wave per row, f16 output
// ---------------------------------------------------------------------------
__global__ __launch_bounds__(256) void ln_kernel(
    const float* __restrict__ x, const float* __restrict__ w,
    const float* __restrict__ b, _Float16* __restrict__ out, int nrows) {
  const int wave = threadIdx.x >> 5, l = threadIdx.x & 31;
  const int row = blockIdx.x * 8 + wave;
  if (row >= nrows) return;
  const float* xr = x + (size_t)row * CM;
  float vals[8], s = 0.f, s2 = 0.f;
#pragma unroll
  for (int j = 0; j < 8; ++j) {
    const float t = xr[j * 32 + l];
    vals[j] = t; s += t; s2 += t * t;
  }
#pragma unroll
  for (int d = 1; d < 32; d <<= 1) {
    s += __shfl_xor(s, d, 32);
    s2 += __shfl_xor(s2, d, 32);
  }
  const float mu = s * (1.0f / CM);
  const float rs = rsqrtf(s2 * (1.0f / CM) - mu * mu + 1e-5f);
  _Float16* orow = out + (size_t)row * CM;
#pragma unroll
  for (int j = 0; j < 8; ++j) {
    const int c = j * 32 + l;
    orow[c] = (_Float16)((vals[j] - mu) * rs * w[c] + b[c]);
  }
}

// ---------------------------------------------------------------------------
// pair bias: pb[h][r1][r2] = (LN(z[r1,r2,:]) @ wb)[h]; one wave per (r1,r2)
// ---------------------------------------------------------------------------
__global__ __launch_bounds__(256) void pairbias_kernel(
    const float* __restrict__ z, const float* __restrict__ w,
    const float* __restrict__ b, const float* __restrict__ wb,
    float* __restrict__ pb) {
  const int wave = threadIdx.x >> 5, l = threadIdx.x & 31;
  const int row = blockIdx.x * 8 + wave;  // r1*R + r2
  const float* zr = z + (size_t)row * CZ;
  float vals[4], s = 0.f, s2 = 0.f;
#pragma unroll
  for (int j = 0; j < 4; ++j) {
    const float t = zr[j * 32 + l];
    vals[j] = t; s += t; s2 += t * t;
  }
#pragma unroll
  for (int d = 1; d < 32; d <<= 1) {
    s += __shfl_xor(s, d, 32);
    s2 += __shfl_xor(s2, d, 32);
  }
  const float mu = s * (1.0f / CZ);
  const float rs = rsqrtf(s2 * (1.0f / CZ) - mu * mu + 1e-5f);
  float acc[8] = {};
#pragma unroll
  for (int j = 0; j < 4; ++j) {
    const int c = j * 32 + l;
    const float xn = (vals[j] - mu) * rs * w[c] + b[c];
#pragma unroll
    for (int hh = 0; hh < 8; ++hh) acc[hh] += xn * wb[c * H + hh];
  }
#pragma unroll
  for (int hh = 0; hh < 8; ++hh)
#pragma unroll
    for (int d = 1; d < 32; d <<= 1) acc[hh] += __shfl_xor(acc[hh], d, 32);
  if (l == 0) {
#pragma unroll
    for (int hh = 0; hh < 8; ++hh) pb[(size_t)hh * R * R + row] = acc[hh];
  }
}

// convert f32 weight [K][N] -> f16 transposed [N][K] (one-time per call)
__global__ __launch_bounds__(256) void cvtT_kernel(
    const float* __restrict__ in, _Float16* __restrict__ out, int N, int K) {
  const int idx = blockIdx.x * 256 + threadIdx.x;
  if (idx < N * K) {
    const int n = idx / K, kk = idx - n * K;
    out[idx] = (_Float16)in[(size_t)kk * N + n];
  }
}

// ---------------------------------------------------------------------------
extern "C" void kernel_launch(void* const* d_in, const int* in_sizes, int n_in,
                              void* d_out, int out_size, void* d_ws, size_t ws_size,
                              hipStream_t stream) {
  (void)in_sizes; (void)n_in; (void)out_size; (void)ws_size;
  const float* m        = (const float*)d_in[0];
  const float* m_mask   = (const float*)d_in[1];
  const float* z        = (const float*)d_in[2];
  const float* row_ln_w = (const float*)d_in[3];
  const float* row_ln_b = (const float*)d_in[4];
  const float* row_z_ln_w = (const float*)d_in[5];
  const float* row_z_ln_b = (const float*)d_in[6];
  const float* row_wq = (const float*)d_in[7];
  const float* row_wk = (const float*)d_in[8];
  const float* row_wv = (const float*)d_in[9];
  const float* row_wb = (const float*)d_in[10];
  const float* row_wg = (const float*)d_in[11];
  const float* row_bg = (const float*)d_in[12];
  const float* row_wo = (const float*)d_in[13];
  const float* row_bo = (const float*)d_in[14];
  const float* col_ln_w = (const float*)d_in[15];
  const float* col_ln_b = (const float*)d_in[16];
  const float* col_wq = (const float*)d_in[17];
  const float* col_wk = (const float*)d_in[18];
  const float* col_wv = (const float*)d_in[19];
  const float* col_wg = (const float*)d_in[20];
  const float* col_bg = (const float*)d_in[21];
  const float* col_wo = (const float*)d_in[22];
  const float* col_bo = (const float*)d_in[23];
  const float* tr_ln_w = (const float*)d_in[24];
  const float* tr_ln_b = (const float*)d_in[25];
  const float* tr_w1 = (const float*)d_in[26];
  const float* tr_b1 = (const float*)d_in[27];
  const float* tr_w2 = (const float*)d_in[28];
  const float* tr_b2 = (const float*)d_in[29];

  // workspace carve-out
  char* ws = (char*)d_ws;
  size_t off = 0;
  auto alloc = [&](size_t bytes) -> char* {
    char* p = ws + off;
    off += (bytes + 255) & ~(size_t)255;
    return p;
  };
  _Float16* xln  = (_Float16*)alloc((size_t)SR * CM * 2);
  _Float16* qkvg = (_Float16*)alloc((size_t)SR * HC * 2 * 4);
  _Float16* qb = qkvg;
  _Float16* kb = qb + (size_t)SR * HC;
  _Float16* vb = kb + (size_t)SR * HC;
  _Float16* gb = vb + (size_t)SR * HC;
  _Float16* t1 = qkvg;  // [SR, CM4] reuse: q/k/v/g dead by transition time
  _Float16* ob = (_Float16*)alloc((size_t)SR * HC * 2);
  float*    pb = (float*)alloc((size_t)H * R * R * 4);
  _Float16* w16 = (_Float16*)alloc((size_t)(10 * CM * HC + 2 * CM * CM4) * 2);
  _Float16* rwq = w16;                       // all stored TRANSPOSED [N][K]
  _Float16* rwk = rwq + CM * HC;
  _Float16* rwv = rwk + CM * HC;
  _Float16* rwg = rwv + CM * HC;
  _Float16* rwo = rwg + CM * HC;
  _Float16* cwq = rwo + HC * CM;
  _Float16* cwk = cwq + CM * HC;
  _Float16* cwv = cwk + CM * HC;
  _Float16* cwg = cwv + CM * HC;
  _Float16* cwo = cwg + CM * HC;
  _Float16* tw1 = cwo + HC * CM;
  _Float16* tw2 = tw1 + (size_t)CM * CM4;

  auto cvtT = [&](const float* src, _Float16* dst, int N, int K) {
    cvtT_kernel<<<(N * K + 255) / 256, 256, 0, stream>>>(src, dst, N, K);
  };
  cvtT(row_wq, rwq, HC, CM); cvtT(row_wk, rwk, HC, CM);
  cvtT(row_wv, rwv, HC, CM); cvtT(row_wg, rwg, HC, CM);
  cvtT(row_wo, rwo, CM, HC);
  cvtT(col_wq, cwq, HC, CM); cvtT(col_wk, cwk, HC, CM);
  cvtT(col_wv, cwv, HC, CM); cvtT(col_wg, cwg, HC, CM);
  cvtT(col_wo, cwo, CM, HC);
  cvtT(tr_w1, tw1, CM4, CM); cvtT(tr_w2, tw2, CM, CM4);

  float* mout = (float*)d_out;
  hipMemcpyAsync(mout, m, (size_t)SR * CM * 4, hipMemcpyDeviceToDevice, stream);
  const float qscale = 0.17677669529663687f;  // CA^-0.5

  const dim3 gProj(HC / 128, SR / 128);   // N=256 GEMMs
  const dim3 gOut(CM / 128, SR / 128);    // output projections
  const dim3 gT1(CM4 / 128, SR / 128);    // transition up-proj

  // ---- pair bias ----
  pairbias_kernel<<<(R * R) / 8, 256, 0, stream>>>(z, row_z_ln_w, row_z_ln_b, row_wb, pb);

  // ---- row attention ----
  ln_kernel<<<SR / 8, 256, 0, stream>>>(mout, row_ln_w, row_ln_b, xln, SR);
  gemm_kernel<0, true, false, false><<<gProj, 256, 0, stream>>>(xln, rwq, nullptr, nullptr, qb, SR, HC, CM, qscale);
  gemm_kernel<0, true, false, false><<<gProj, 256, 0, stream>>>(xln, rwk, nullptr, nullptr, kb, SR, HC, CM, 1.0f);
  gemm_kernel<0, true, false, false><<<gProj, 256, 0, stream>>>(xln, rwv, nullptr, nullptr, vb, SR, HC, CM, 1.0f);
  gemm_kernel<1, true, false, true><<<gProj, 256, 0, stream>>>(xln, rwg, row_bg, nullptr, gb, SR, HC, CM, 1.0f);
  atten_kernel<false><<<dim3(S, H), 256, 0, stream>>>(qb, kb, vb, gb, pb, m_mask, ob);
  gemm_kernel<0, false, true, true><<<gOut, 256, 0, stream>>>(ob, rwo, row_bo, mout, nullptr, SR, CM, HC, 1.0f);

  // ---- column attention ----
  ln_kernel<<<SR / 8, 256, 0, stream>>>(mout, col_ln_w, col_ln_b, xln, SR);
  gemm_kernel<0, true, false, false><<<gProj, 256, 0, stream>>>(xln, cwq, nullptr, nullptr, qb, SR, HC, CM, qscale);
  gemm_kernel<0, true, false, false><<<gProj, 256, 0, stream>>>(xln, cwk, nullptr, nullptr, kb, SR, HC, CM, 1.0f);
  gemm_kernel<0, true, false, false><<<gProj, 256, 0, stream>>>(xln, cwv, nullptr, nullptr, vb, SR, HC, CM, 1.0f);
  gemm_kernel<1, true, false, true><<<gProj, 256, 0, stream>>>(xln, cwg, col_bg, nullptr, gb, SR, HC, CM, 1.0f);
  atten_kernel<true><<<dim3(R, H), 256, 0, stream>>>(qb, kb, vb, gb, nullptr, m_mask, ob);
  gemm_kernel<0, false, true, true><<<gOut, 256, 0, stream>>>(ob, cwo, col_bo, mout, nullptr, SR, CM, HC, 1.0f);

  // ---- transition ----
  ln_kernel<<<SR / 8, 256, 0, stream>>>(mout, tr_ln_w, tr_ln_b, xln, SR);
  gemm_kernel<2, true, false, true><<<gT1, 256, 0, stream>>>(xln, tw1, tr_b1, nullptr, t1, SR, CM4, CM, 1.0f);
  gemm_kernel<0, false, true, true><<<gOut, 256, 0, stream>>>(t1, tw2, tr_b2, mout, nullptr, SR, CM, CM4, 1.0f);
}
